// GCPNPolicy_23235773071826
// MI455X (gfx1250) — matrix operand order repaired and checked by reading, not compiled, not run on hardware
//
#include <hip/hip_runtime.h>

#define B_GRAPHS 4096
#define NTOK 9
#define NEGV (-1e30f)

typedef __attribute__((ext_vector_type(16))) __bf16 v16bf;
typedef __attribute__((ext_vector_type(8)))  float  v8f;

enum { M_NODE1 = 0, M_NODE2 = 1, M_EDGE = 2, M_GF = 3, M_PLAIN = 4 };

struct GemmParams {
  const __bf16* nf;      // node_feature bf16 [N x 128]
  const __bf16* gf;      // graph_feature bf16 [B x 256]
  const __bf16* ae;      // atom_embeddings bf16 [9 x 128]
  const __bf16* Xplain;  // row-major [M x 256] (layer-2 input)
  const __bf16* W;       // [Kin x 256] bf16 row-major (global; staged to LDS)
  const float*  b0;      // [256]
  const float*  W2;      // [256 x O] fp32 (layer-2 only)
  const float*  b2;      // [O]
  __bf16* H1out;         // layer-1 output [M x 256] bf16
  float*  Lout;          // layer-2 output [M x O] fp32
  const int* n2g;        // node2graph [N]
  const int* n1idx;      // [B]
  const int* n2idx;      // [B]
  int M;                 // rows
  int Nnodes;            // N
  int Kin;               // 256 / 384 / 512
};

// Compile-time K-region table (in ksteps of 32) per input-assembly mode.
template <int MODE> struct RegCfg;
template <> struct RegCfg<M_NODE1> { static constexpr int n = 2; static constexpr int b[3] = {0, 4, 12}; };
template <> struct RegCfg<M_NODE2> { static constexpr int n = 3; static constexpr int b[4] = {0, 4, 12, 16}; };
template <> struct RegCfg<M_EDGE>  { static constexpr int n = 2; static constexpr int b[3] = {0, 4, 8}; };
template <> struct RegCfg<M_GF>    { static constexpr int n = 1; static constexpr int b[2] = {0, 8}; };
template <> struct RegCfg<M_PLAIN> { static constexpr int n = 1; static constexpr int b[2] = {0, 8}; };

__device__ __forceinline__ const __bf16* ext_row_ptr(const GemmParams& p, int r) {
  if (r < p.Nnodes) return p.nf + (size_t)r * 128;
  return p.ae + (size_t)((r - p.Nnodes) % NTOK) * 128;
}

// Fill per-strip region base pointers (one gather per strip, not per kstep).
template <int MODE>
__device__ __forceinline__ void setup_regions(const GemmParams& p, int row,
                                              const __bf16** rbase) {
  if constexpr (MODE == M_NODE1) {
    rbase[0] = p.nf + (size_t)row * 128;
    rbase[1] = p.gf + (size_t)p.n2g[row] * 256;
  } else if constexpr (MODE == M_NODE2) {
    rbase[0] = ext_row_ptr(p, row);
    int g = (row < p.Nnodes) ? p.n2g[row] : (row - p.Nnodes) / NTOK;
    rbase[1] = p.gf + (size_t)g * 256;
    rbase[2] = p.nf + (size_t)p.n1idx[g] * 128;
  } else if constexpr (MODE == M_EDGE) {
    rbase[0] = ext_row_ptr(p, p.n1idx[row]);
    rbase[1] = ext_row_ptr(p, p.n2idx[row]);
  } else if constexpr (MODE == M_GF) {
    rbase[0] = p.gf + (size_t)row * 256;
  } else {
    rbase[0] = p.Xplain + (size_t)row * 256;
  }
}

// A fragment (16-bit 16x32 layout), ap already offset by half*8:
// elements 0..7  <-> K = k0 + half*8 .. +7       (contiguous 16B)
// elements 8..15 <-> K = k0 + 16 + half*8 .. +7  (contiguous 16B)
__device__ __forceinline__ v16bf load_a_frag(const __bf16* ap) {
  v16bf a;
#pragma unroll
  for (int j = 0; j < 8; ++j) { a[j] = ap[j]; a[8 + j] = ap[16 + j]; }
  return a;
}

__device__ __forceinline__ v16bf load_frag16(const __bf16* p) {
  v16bf v;
#pragma unroll
  for (int i = 0; i < 16; ++i) v[i] = p[i];
  return v;
}

// One wave owns a 16-row strip; 2 passes over 8 col-tiles, K in branch-free
// per-region steps of 32.
// O == 0 : layer-1, write relu(X@W+b) as bf16.
// O  > 0 : layer-2, fuse relu + dot with W2[256xO] + lane reduction -> logits.
template <int MODE, int O>
__global__ __launch_bounds__(256) void gemm_kernel(GemmParams p) {
  extern __shared__ char smem[];
  __bf16* Ws = (__bf16*)smem;                          // Kin*256, B-fragment swizzled
  float*  bs = (float*)(smem + (size_t)p.Kin * 512);   // 256
  float*  w2s = bs + 256;                              // 256*O
  float*  b2s = w2s + 256 * (O > 0 ? O : 0);

  const int tid = threadIdx.x;
  const int total = p.Kin * 256;
  // Pre-swizzle W into B-fragment order: [ks][ct][lane][e],
  // elem e <-> K = ks*32 + (e>>3)*16 + (lane>>4)*8 + (e&7), N = ct*16 + (lane&15)
  for (int idx = tid; idx < total; idx += 256) {
    int e = idx & 15, ln = (idx >> 4) & 31, ct = (idx >> 9) & 15, ks = idx >> 13;
    int k = ks * 32 + (e >> 3) * 16 + (ln >> 4) * 8 + (e & 7);
    Ws[idx] = p.W[(size_t)k * 256 + ct * 16 + (ln & 15)];
  }
  for (int i = tid; i < 256; i += 256) bs[i] = p.b0[i];
  if (O > 0) {
    for (int i = tid; i < 256 * O; i += 256) w2s[i] = p.W2[i];
    if (tid < O) b2s[tid] = p.b2[tid];
  }
  __syncthreads();

  const int lane = tid & 31, wave = tid >> 5;
  const int half = lane >> 4, lm = lane & 15;
  const int nstrips = (p.M + 15) >> 4;
  constexpr int NREG = RegCfg<MODE>::n;

  for (int strip = blockIdx.x * 8 + wave; strip < nstrips; strip += gridDim.x * 8) {
    const int row0 = strip << 4;
    int arow = row0 + lm;
    if (arow >= p.M) arow = p.M - 1;  // clamped load; invalid rows never stored

    const __bf16* rbase[NREG];
    setup_regions<MODE>(p, arow, rbase);

    // Prefetch next strip's first activation region (global_prefetch_b8).
    {
      int nstrip = strip + gridDim.x * 8;
      if (nstrip < nstrips) {
        int nrow = (nstrip << 4) + lm;
        if (nrow >= p.M) nrow = p.M - 1;
        const __bf16* npf[NREG];
        setup_regions<MODE>(p, nrow, npf);
        __builtin_prefetch(npf[0], 0, 0);
        __builtin_prefetch(npf[0] + 64, 0, 0);
      }
    }

    float facc[(O > 0) ? 8 * O : 1];
    if (O > 0) {
#pragma unroll
      for (int i = 0; i < 8 * O; ++i) facc[i] = 0.f;
    }

#pragma unroll
    for (int ctb = 0; ctb < 16; ctb += 8) {
      v8f cacc[8] = {};
#pragma unroll
      for (int rg = 0; rg < NREG; ++rg) {
        const __bf16* ap = rbase[rg] + half * 8;
#pragma unroll
        for (int s = RegCfg<MODE>::b[rg]; s < RegCfg<MODE>::b[rg + 1]; ++s) {
          v16bf a = load_a_frag(ap);
          ap += 32;
          const __bf16* bp = Ws + ((size_t)(s * 16 + ctb) * 32 + lane) * 16;
#pragma unroll
          for (int t = 0; t < 8; ++t) {
            v16bf bf = load_frag16(bp + (size_t)t * 512);
            cacc[t] = __builtin_amdgcn_wmma_f32_16x16x32_bf16(
                false, a, false, bf, (short)0, cacc[t], false, false);
          }
        }
      }
#pragma unroll
      for (int t = 0; t < 8; ++t) {
        const int col = (ctb + t) * 16 + lm;
#pragma unroll
        for (int r = 0; r < 8; ++r) {
          const int rr = row0 + half * 8 + r;  // C layout: VGPR r -> M = half*8+r, N = lane&15
          float v = cacc[t][r] + bs[col];
          v = v > 0.f ? v : 0.f;
          if (O == 0) {
            if (rr < p.M) p.H1out[(size_t)rr * 256 + col] = (__bf16)v;
          } else {
#pragma unroll
            for (int o = 0; o < O; ++o) facc[r * O + o] += v * w2s[col * O + o];
          }
        }
      }
    }

    if (O > 0) {
#pragma unroll
      for (int i = 0; i < 8 * O; ++i) {
        float v = facc[i];
        v += __shfl_xor(v, 1, 32); v += __shfl_xor(v, 2, 32);
        v += __shfl_xor(v, 4, 32); v += __shfl_xor(v, 8, 32);
        facc[i] = v;
      }
      if (lm == 0) {
#pragma unroll
        for (int r = 0; r < 8; ++r) {
          int rr = row0 + half * 8 + r;
          if (rr < p.M) {
#pragma unroll
            for (int o = 0; o < O; ++o) p.Lout[(size_t)rr * O + o] = facc[r * O + o] + b2s[o];
          }
        }
      }
    }
  }
}

__global__ void cvt_bf16_kernel(const float* __restrict__ src, __bf16* __restrict__ dst, int n) {
  for (int i = blockIdx.x * blockDim.x + threadIdx.x; i < n; i += gridDim.x * blockDim.x)
    dst[i] = (__bf16)src[i];
}

__global__ void prep_kernel(const int* __restrict__ num_nodes, const int* __restrict__ actions,
                            int* __restrict__ offsets, int* __restrict__ n1idx,
                            int* __restrict__ n2idx, int Nn) {
  __shared__ int part[256];
  __shared__ int base[256];
  const int tid = threadIdx.x;
  int s = 0;
  for (int i = 0; i < 16; ++i) s += num_nodes[tid * 16 + i];
  part[tid] = s;
  __syncthreads();
  if (tid == 0) {
    int acc = 0;
    for (int i = 0; i < 256; ++i) { base[i] = acc; acc += part[i]; }
  }
  __syncthreads();
  int acc = base[tid];
  for (int i = 0; i < 16; ++i) {
    int b = tid * 16 + i;
    offsets[b] = acc;
    acc += num_nodes[b];
  }
  __syncthreads();
  for (int b = tid; b < B_GRAPHS; b += 256) {
    int nn = num_nodes[b], off = offsets[b];
    int a1 = actions[b * 4 + 0], a2 = actions[b * 4 + 1];
    n1idx[b] = off + a1;
    n2idx[b] = (a2 < nn) ? (off + a2) : (Nn + b * NTOK + (a2 - nn));
  }
}

__global__ void mask_kernel(float* __restrict__ l2, const int* __restrict__ src,
                            const int* __restrict__ dst, const int* __restrict__ n2g,
                            const int* __restrict__ n1idx, int E) {
  const int gid = blockIdx.x * blockDim.x + threadIdx.x;
  if (gid < B_GRAPHS) l2[n1idx[gid]] = NEGV;
  const int stride = gridDim.x * blockDim.x;
  for (int e = gid; e < E; e += stride) {
    int s = src[e];
    if (s == n1idx[n2g[s]]) l2[dst[e]] = NEGV;  // all writes are NEGV: race-free
  }
}

__global__ void seg1_kernel(const float* __restrict__ l1, const int* __restrict__ offsets,
                            const int* __restrict__ num_nodes, const int* __restrict__ actions,
                            float* __restrict__ lp_a, float* __restrict__ ent) {
  const int wave = threadIdx.x >> 5, lane = threadIdx.x & 31;
  const int b = blockIdx.x * 8 + wave;
  if (b >= B_GRAPHS) return;
  const int nn = num_nodes[b], off = offsets[b];
  float m = -3.0e38f;
  for (int i = lane; i < nn; i += 32) m = fmaxf(m, l1[off + i]);
  for (int d = 16; d; d >>= 1) m = fmaxf(m, __shfl_xor(m, d, 32));
  float s = 0.f, t = 0.f;
  for (int i = lane; i < nn; i += 32) {
    float z = l1[off + i] - m;
    float e = __expf(z);
    s += e; t += z * e;
  }
  for (int d = 16; d; d >>= 1) { s += __shfl_xor(s, d, 32); t += __shfl_xor(t, d, 32); }
  if (lane == 0) {
    float ls = __logf(s);
    lp_a[b] = (l1[off + actions[b * 4 + 0]] - m) - ls;
    ent[b] = ls - t / s;
  }
}

__global__ void seg2_kernel(const float* __restrict__ l2, const int* __restrict__ offsets,
                            const int* __restrict__ num_nodes, const int* __restrict__ n2idx,
                            float* __restrict__ lp_a, float* __restrict__ ent, int Nn) {
  const int wave = threadIdx.x >> 5, lane = threadIdx.x & 31;
  const int b = blockIdx.x * 8 + wave;
  if (b >= B_GRAPHS) return;
  const int nn = num_nodes[b], off = offsets[b];
  const int cnt = nn + NTOK;
  float m = -3.0e38f;
  for (int i = lane; i < cnt; i += 32) {
    int r = (i < nn) ? (off + i) : (Nn + b * NTOK + (i - nn));
    m = fmaxf(m, l2[r]);
  }
  for (int d = 16; d; d >>= 1) m = fmaxf(m, __shfl_xor(m, d, 32));
  float s = 0.f, t = 0.f;
  for (int i = lane; i < cnt; i += 32) {
    int r = (i < nn) ? (off + i) : (Nn + b * NTOK + (i - nn));
    float z = l2[r] - m;
    float e = __expf(z);
    s += e; t += z * e;
  }
  for (int d = 16; d; d >>= 1) { s += __shfl_xor(s, d, 32); t += __shfl_xor(t, d, 32); }
  if (lane == 0) {
    float ls = __logf(s);
    lp_a[b] = (l2[n2idx[b]] - m) - ls;
    ent[b] = ls - t / s;
  }
}

__global__ void final_kernel(const float* __restrict__ lE, const float* __restrict__ lS,
                             const float* __restrict__ lC, const float* __restrict__ lp1a,
                             const float* __restrict__ ent1, const float* __restrict__ lp2a,
                             const float* __restrict__ ent2, const int* __restrict__ actions,
                             float* __restrict__ out) {
  const int b = blockIdx.x * blockDim.x + threadIdx.x;
  if (b >= B_GRAPHS) return;
  // edge: 3-way log-softmax
  float e0 = lE[b * 3 + 0], e1 = lE[b * 3 + 1], e2 = lE[b * 3 + 2];
  float m = fmaxf(e0, fmaxf(e1, e2));
  float x0 = __expf(e0 - m), x1 = __expf(e1 - m), x2 = __expf(e2 - m);
  float s = x0 + x1 + x2;
  float t = (e0 - m) * x0 + (e1 - m) * x1 + (e2 - m) * x2;
  float ls = __logf(s);
  int a3 = actions[b * 4 + 2];
  float lpe_a = ((a3 == 0 ? e0 : (a3 == 1 ? e1 : e2)) - m) - ls;
  float ent3 = ls - t / s;
  // stop: 2-way log-softmax
  float s0 = lS[b * 2 + 0], s1 = lS[b * 2 + 1];
  float ms = fmaxf(s0, s1);
  float y0 = __expf(s0 - ms), y1 = __expf(s1 - ms);
  float ss = y0 + y1;
  float ts = (s0 - ms) * y0 + (s1 - ms) * y1;
  float lss = __logf(ss);
  int a4 = actions[b * 4 + 3];
  float lps_a = ((a4 == 0 ? s0 : s1) - ms) - lss;
  float ent4 = lss - ts / ss;

  out[b] = lC[b];
  out[B_GRAPHS + b] = lp1a[b] + lp2a[b] + lpe_a + lps_a;
  out[2 * B_GRAPHS + b] = ent1[b] + ent2[b] + ent3 + ent4;
}

extern "C" void kernel_launch(void* const* d_in, const int* in_sizes, int n_in,
                              void* d_out, int out_size, void* d_ws, size_t ws_size,
                              hipStream_t stream) {
  (void)n_in; (void)out_size; (void)ws_size;
  const float* node_feature  = (const float*)d_in[0];
  const float* graph_feature = (const float*)d_in[1];
  const float* atom_emb      = (const float*)d_in[2];
  const int* node2graph = (const int*)d_in[33];
  const int* num_nodes  = (const int*)d_in[34];
  const int* edge_index = (const int*)d_in[35];
  const int* actions    = (const int*)d_in[36];
  const int N    = in_sizes[0] / 128;
  const int E    = in_sizes[35] / 2;
  const int Next = N + B_GRAPHS * NTOK;

  char* wsb = (char*)d_ws;
  size_t off = 0;
  auto alloc = [&](size_t bytes) -> char* {
    char* p = wsb + off;
    off = (off + bytes + 255) & ~((size_t)255);
    return p;
  };
  int* offsets = (int*)alloc((size_t)B_GRAPHS * 4);
  int* n1idx   = (int*)alloc((size_t)B_GRAPHS * 4);
  int* n2idx   = (int*)alloc((size_t)B_GRAPHS * 4);
  __bf16* nf16 = (__bf16*)alloc((size_t)N * 128 * 2);
  __bf16* gf16 = (__bf16*)alloc((size_t)B_GRAPHS * 256 * 2);
  __bf16* ae16 = (__bf16*)alloc((size_t)NTOK * 128 * 2);
  const int wIdx[10] = {3, 5, 9, 11, 15, 17, 21, 23, 27, 29};  // hidden-layer W's
  __bf16* w16[10];
  for (int i = 0; i < 10; ++i) w16[i] = (__bf16*)alloc((size_t)in_sizes[wIdx[i]] * 2);
  __bf16* H1 = (__bf16*)alloc((size_t)Next * 256 * 2);  // reused by every MLP sequentially
  float* l1  = (float*)alloc((size_t)N * 4);
  float* l2  = (float*)alloc((size_t)Next * 4);
  float* lE  = (float*)alloc((size_t)B_GRAPHS * 3 * 4);
  float* lS  = (float*)alloc((size_t)B_GRAPHS * 2 * 4);
  float* lC  = (float*)alloc((size_t)B_GRAPHS * 4);
  float* lp1a = (float*)alloc((size_t)B_GRAPHS * 4);
  float* ent1 = (float*)alloc((size_t)B_GRAPHS * 4);
  float* lp2a = (float*)alloc((size_t)B_GRAPHS * 4);
  float* ent2 = (float*)alloc((size_t)B_GRAPHS * 4);

  prep_kernel<<<1, 256, 0, stream>>>(num_nodes, actions, offsets, n1idx, n2idx, N);

  auto cvt = [&](const float* s, __bf16* d, int n) {
    int blocks = (n + 255) / 256; if (blocks > 2048) blocks = 2048;
    cvt_bf16_kernel<<<blocks, 256, 0, stream>>>(s, d, n);
  };
  cvt(node_feature, nf16, N * 128);
  cvt(graph_feature, gf16, B_GRAPHS * 256);
  cvt(atom_emb, ae16, NTOK * 128);
  for (int i = 0; i < 10; ++i) cvt((const float*)d_in[wIdx[i]], w16[i], in_sizes[wIdx[i]]);

  GemmParams gp{};
  gp.nf = nf16; gp.gf = gf16; gp.ae = ae16;
  gp.n2g = node2graph; gp.n1idx = n1idx; gp.n2idx = n2idx;
  gp.Nnodes = N;

  auto ggrid = [](int M) { int ns = (M + 15) / 16; int g = (ns + 7) / 8;
                           if (g > 2048) g = 2048; if (g < 1) g = 1; return g; };
  auto shm = [](int Kin, int O) { return (size_t)Kin * 512 + 1024 +
                                         (O > 0 ? (size_t)256 * O * 4 + 64 : (size_t)0); };

  // ---- p_node1 ----
  { GemmParams p = gp; p.M = N; p.Kin = 384; p.W = w16[0]; p.b0 = (const float*)d_in[4]; p.H1out = H1;
    gemm_kernel<M_NODE1, 0><<<ggrid(N), 256, shm(384, 0), stream>>>(p); }
  { GemmParams p = gp; p.M = N; p.Kin = 256; p.W = w16[1]; p.b0 = (const float*)d_in[6];
    p.Xplain = H1; p.W2 = (const float*)d_in[7]; p.b2 = (const float*)d_in[8]; p.Lout = l1;
    gemm_kernel<M_PLAIN, 1><<<ggrid(N), 256, shm(256, 1), stream>>>(p); }
  seg1_kernel<<<B_GRAPHS / 8, 256, 0, stream>>>(l1, offsets, num_nodes, actions, lp1a, ent1);

  // ---- p_node2 ----
  { GemmParams p = gp; p.M = Next; p.Kin = 512; p.W = w16[2]; p.b0 = (const float*)d_in[10]; p.H1out = H1;
    gemm_kernel<M_NODE2, 0><<<ggrid(Next), 256, shm(512, 0), stream>>>(p); }
  { GemmParams p = gp; p.M = Next; p.Kin = 256; p.W = w16[3]; p.b0 = (const float*)d_in[12];
    p.Xplain = H1; p.W2 = (const float*)d_in[13]; p.b2 = (const float*)d_in[14]; p.Lout = l2;
    gemm_kernel<M_PLAIN, 1><<<ggrid(Next), 256, shm(256, 1), stream>>>(p); }
  { int blocks = (E + 255) / 256; if (blocks > 4096) blocks = 4096;
    mask_kernel<<<blocks, 256, 0, stream>>>(l2, edge_index, edge_index + E, node2graph, n1idx, E); }
  seg2_kernel<<<B_GRAPHS / 8, 256, 0, stream>>>(l2, offsets, num_nodes, n2idx, lp2a, ent2, N);

  // ---- p_edge ----
  { GemmParams p = gp; p.M = B_GRAPHS; p.Kin = 256; p.W = w16[4]; p.b0 = (const float*)d_in[16]; p.H1out = H1;
    gemm_kernel<M_EDGE, 0><<<ggrid(B_GRAPHS), 256, shm(256, 0), stream>>>(p); }
  { GemmParams p = gp; p.M = B_GRAPHS; p.Kin = 256; p.W = w16[5]; p.b0 = (const float*)d_in[18];
    p.Xplain = H1; p.W2 = (const float*)d_in[19]; p.b2 = (const float*)d_in[20]; p.Lout = lE;
    gemm_kernel<M_PLAIN, 3><<<ggrid(B_GRAPHS), 256, shm(256, 3), stream>>>(p); }

  // ---- p_stop ----
  { GemmParams p = gp; p.M = B_GRAPHS; p.Kin = 256; p.W = w16[6]; p.b0 = (const float*)d_in[22]; p.H1out = H1;
    gemm_kernel<M_GF, 0><<<ggrid(B_GRAPHS), 256, shm(256, 0), stream>>>(p); }
  { GemmParams p = gp; p.M = B_GRAPHS; p.Kin = 256; p.W = w16[7]; p.b0 = (const float*)d_in[24];
    p.Xplain = H1; p.W2 = (const float*)d_in[25]; p.b2 = (const float*)d_in[26]; p.Lout = lS;
    gemm_kernel<M_PLAIN, 2><<<ggrid(B_GRAPHS), 256, shm(256, 2), stream>>>(p); }

  // ---- p_critic ----
  { GemmParams p = gp; p.M = B_GRAPHS; p.Kin = 256; p.W = w16[8]; p.b0 = (const float*)d_in[28]; p.H1out = H1;
    gemm_kernel<M_GF, 0><<<ggrid(B_GRAPHS), 256, shm(256, 0), stream>>>(p); }
  { GemmParams p = gp; p.M = B_GRAPHS; p.Kin = 256; p.W = w16[9]; p.b0 = (const float*)d_in[30];
    p.Xplain = H1; p.W2 = (const float*)d_in[31]; p.b2 = (const float*)d_in[32]; p.Lout = lC;
    gemm_kernel<M_PLAIN, 1><<<ggrid(B_GRAPHS), 256, shm(256, 1), stream>>>(p); }

  final_kernel<<<(B_GRAPHS + 255) / 256, 256, 0, stream>>>(
      lE, lS, lC, lp1a, ent1, lp2a, ent2, actions, (float*)d_out);
}
